// TransformerBlock_85117661872730
// MI455X (gfx1250) — matrix operand order repaired and checked by reading, not compiled, not run on hardware
//
#include <hip/hip_runtime.h>
#include <cstdint>

// ---------------------------------------------------------------------------
// Transformer block (pre-norm, causal MHA, GELU FFN) for MI455X / gfx1250.
// fp32 -> bf16 operands, f32 accumulation via v_wmma_f32_16x16x32_bf16.
// GEMMs: block-tiled 128x128, K-slabs of 32 staged into LDS by the Tensor
// Data Mover (tensor_load_to_lds, TENSORcnt), double-buffered; WMMA feeds
// from ds_load_b128. Attention: block-cooperative flash attention -- K/V
// chunks TDM-staged into shared LDS (8 waves reuse them), online softmax,
// P staged through per-wave LDS, V stored transposed.
// ---------------------------------------------------------------------------

#define D_MODEL 768
#define N_HEADS 12
#define D_HEAD  64
#define D_FF    3072
#define SEQ_L   2048
#define BATCH   2
#define M_TOT   (BATCH * SEQ_L)   // 4096 tokens
#define LN_EPS  1e-5f

// GEMM LDS tile: 128 rows x 32 bf16, TDM-padded 64B row -> 80B stride (ld 40).
#define TILE_K    32
#define TILE_MN   128
#define LDS_LD    40                       // elements (80 bytes)
#define LDS_TILE  (TILE_MN * LDS_LD)       // 5120 elements = 10KB
// Attention K tile: 32 key-rows x 64 d, padded 128B row -> 144B stride (ld 72).
#define KT_LD     72
// Attention V^T tile: 64 d-rows x 32 keys, padded 64B row -> 80B stride (ld 40).
#define VT_LD     40

typedef __attribute__((ext_vector_type(16))) __bf16 v16bf;
typedef __attribute__((ext_vector_type(8)))  float  v8f;
typedef __attribute__((ext_vector_type(4)))  unsigned int v4u;
typedef __attribute__((ext_vector_type(8)))  int v8i;
typedef __attribute__((ext_vector_type(4)))  int v4i;

union FragU { v16bf v; uint4 q[2]; };

// float -> bf16 round-to-nearest-even, raw u16.
__device__ __forceinline__ unsigned short f2bf(float x) {
  union { float f; unsigned int u; } t; t.f = x;
  unsigned int u = t.u;
  u += 0x7FFFu + ((u >> 16) & 1u);
  return (unsigned short)(u >> 16);
}

// A fragment (16xK, ISA 7.12.2 16-bit A layout): lane m (0..15) row m,
// K = k0+{0..7,16..23}; lanes 16..31: K = k0+{8..15,24..31}.
__device__ __forceinline__ v16bf load_a_frag(const unsigned short* base, int row0,
                                             int ld, int k0, int lane) {
  const int r  = lane & 15;
  const int kk = k0 + ((lane >> 4) << 3);
  const unsigned short* p = base + (size_t)(row0 + r) * (size_t)ld + kk;
  FragU f;
  f.q[0] = *(const uint4*)(p);
  f.q[1] = *(const uint4*)(p + 16);
  return f.v;
}

// B fragment (Kx16): source S row-major [N][K], B[k][n] = S[row0+n][k].
// Lanes 0..15 hold K = k0+{0..7,16..23}, lanes 16..31 hold K = k0+{8..15,24..31}.
__device__ __forceinline__ v16bf load_b_frag(const unsigned short* base, int row0,
                                             int ld, int k0, int lane) {
  const int n  = lane & 15;
  const int kk = k0 + ((lane >> 4) << 3);
  const unsigned short* p = base + (size_t)(row0 + n) * (size_t)ld + kk;
  FragU f;
  f.q[0] = *(const uint4*)(p);
  f.q[1] = *(const uint4*)(p + 16);
  return f.v;
}

__device__ __forceinline__ v8f wmma_bf16(v16bf a, v16bf b, v8f c) {
  return __builtin_amdgcn_wmma_f32_16x16x32_bf16(false, a, false, b, (short)0, c,
                                                 false, false);
}

// ---------------------------------------------------------------------------
// Tensor Data Mover: 2D tile (tile_d0 x tile_d1 elements of 2 bytes) from
// global (row stride d0_stride elements) into LDS at lds_addr.
// LDS padding: pad_amt_code+1 DWORDs inserted after every 2^(pad_int_code+1)
// DWORDs. Descriptor per CDNA5 ISA 08_async_tensor sec. 8.
// ---------------------------------------------------------------------------
__device__ __forceinline__ void tdm_load_2d(unsigned lds_addr, const void* gptr,
                                            unsigned tile_d0, unsigned tile_d1,
                                            unsigned long long d0_stride,
                                            unsigned pad_int_code,
                                            unsigned pad_amt_code) {
  const unsigned long long ga = (unsigned long long)(uintptr_t)gptr;
  v4u g0;
  g0[0] = 1u;                                   // count=1, user mode, no gather
  g0[1] = lds_addr;                             // LDS byte address
  g0[2] = (unsigned)(ga & 0xFFFFFFFFu);         // global addr [95:64]
  g0[3] = (unsigned)((ga >> 32) & 0x01FFFFFFu)  // global addr [120:96] (57b)
          | (2u << 30);                         // type = 2 ("image")
  v8i g1;
  g1[0] = (int)((1u << 16)                      // data_size = 1 -> 2 bytes
          | (1u << 20)                          // pad_enable
          | (pad_int_code << 22)                // pad interval
          | (pad_amt_code << 25));              // pad amount
  g1[1] = (int)((tile_d0 & 0xFFFFu) << 16);     // tensor_dim0[15:0] (== tile)
  g1[2] = (int)(((tile_d0 >> 16) & 0xFFFFu) | ((tile_d1 & 0xFFFFu) << 16));
  g1[3] = (int)(((tile_d1 >> 16) & 0xFFFFu) | ((tile_d0 & 0xFFFFu) << 16)); // tile_dim0
  g1[4] = (int)(tile_d1 & 0xFFFFu);             // tile_dim1; tile_dim2 = 0
  g1[5] = (int)(d0_stride & 0xFFFFFFFFull);     // tensor_dim0_stride[31:0]
  g1[6] = (int)((d0_stride >> 32) & 0xFFFFull); // stride[47:32]; dim1_stride=0
  g1[7] = 0;
  const v4i gz = {0, 0, 0, 0};                  // 2D tile: groups 2/3 unused
#if __clang_major__ >= 23
  const v8i gz8 = {0, 0, 0, 0, 0, 0, 0, 0};
  __builtin_amdgcn_tensor_load_to_lds(g0, g1, gz, gz, gz8, 0);
#else
  __builtin_amdgcn_tensor_load_to_lds(g0, g1, gz, gz, 0);
#endif
}

// ---------------------------------------------------------------------------
// fp32 -> bf16 cast (weights)
// ---------------------------------------------------------------------------
__global__ void cast_bf16_kernel(const float* __restrict__ in,
                                 unsigned short* __restrict__ out, int n) {
  int i = blockIdx.x * blockDim.x + threadIdx.x;
  if (i < n) out[i] = f2bf(in[i]);
}

// ---------------------------------------------------------------------------
// LayerNorm: one 256-thread block per token row; writes bf16 normalized row.
// ---------------------------------------------------------------------------
__global__ void ln_bf16_kernel(const float* __restrict__ x,
                               const float* __restrict__ g,
                               const float* __restrict__ b,
                               unsigned short* __restrict__ out) {
  __shared__ float red[256];
  const int row = blockIdx.x;
  const float* xr = x + (size_t)row * D_MODEL;

  float s = 0.f;
  for (int i = threadIdx.x; i < D_MODEL; i += 256) s += xr[i];
  red[threadIdx.x] = s; __syncthreads();
  for (int off = 128; off > 0; off >>= 1) {
    if (threadIdx.x < off) red[threadIdx.x] += red[threadIdx.x + off];
    __syncthreads();
  }
  const float mu = red[0] / (float)D_MODEL;
  __syncthreads();

  float s2 = 0.f;
  for (int i = threadIdx.x; i < D_MODEL; i += 256) {
    float d = xr[i] - mu; s2 += d * d;
  }
  red[threadIdx.x] = s2; __syncthreads();
  for (int off = 128; off > 0; off >>= 1) {
    if (threadIdx.x < off) red[threadIdx.x] += red[threadIdx.x + off];
    __syncthreads();
  }
  const float rstd = rsqrtf(red[0] / (float)D_MODEL + LN_EPS);

  for (int i = threadIdx.x; i < D_MODEL; i += 256)
    out[(size_t)row * D_MODEL + i] = f2bf((xr[i] - mu) * rstd * g[i] + b[i]);
}

// ---------------------------------------------------------------------------
// WMMA GEMM: C[M,N] = A[M,K](bf16) * W[N,K]^T(bf16) + bias, with epilogues.
// Block = 256 threads = 8 waves -> 128x128 tile (4x2 waves of 32x64).
// K-slabs of 32 staged A/B via TDM into double-buffered LDS.
// ---------------------------------------------------------------------------
enum { EPI_BF16 = 0, EPI_VT = 1, EPI_RES = 2, EPI_GELU = 3 };

template <int EPI>
__global__ void gemm_wmma_kernel(const unsigned short* __restrict__ A,
                                 const unsigned short* __restrict__ B,
                                 const float* __restrict__ bias,
                                 int M, int N, int K,
                                 unsigned short* __restrict__ outB,
                                 const float* __restrict__ res,
                                 float* __restrict__ outF) {
  __shared__ __align__(16) unsigned short ldsA[2][LDS_TILE];
  __shared__ __align__(16) unsigned short ldsB[2][LDS_TILE];

  const int lane = threadIdx.x & 31;
  const int wv   = threadIdx.x >> 5;
  const int m0l  = (wv >> 1) << 5;       // wave's rows within block tile (0..96)
  const int n0l  = (wv & 1) << 6;        // wave's cols within block tile (0/64)
  const int bm   = blockIdx.y;
  const int bn   = blockIdx.x;
  const unsigned short* Ag = A + (size_t)bm * TILE_MN * (size_t)K;
  const unsigned short* Bg = B + (size_t)bn * TILE_MN * (size_t)K;

  // Uniform-branch gate: TDM ignores EXEC, so we must *branch*, not predicate.
  const int wvu = __builtin_amdgcn_readfirstlane((int)(threadIdx.x >> 5));
  const int S = K / TILE_K;

  if (wvu == 0) {
    tdm_load_2d((unsigned)(uintptr_t)&ldsA[0][0], Ag, TILE_K, TILE_MN,
                (unsigned long long)K, 3u, 3u);
    tdm_load_2d((unsigned)(uintptr_t)&ldsB[0][0], Bg, TILE_K, TILE_MN,
                (unsigned long long)K, 3u, 3u);
  }

  v8f acc[2][4] = {};
  for (int s = 0; s < S; ++s) {
    const int buf = s & 1;
    if (wvu == 0) {
      if (s + 1 < S) {
        tdm_load_2d((unsigned)(uintptr_t)&ldsA[buf ^ 1][0],
                    Ag + (size_t)(s + 1) * TILE_K, TILE_K, TILE_MN,
                    (unsigned long long)K, 3u, 3u);
        tdm_load_2d((unsigned)(uintptr_t)&ldsB[buf ^ 1][0],
                    Bg + (size_t)(s + 1) * TILE_K, TILE_K, TILE_MN,
                    (unsigned long long)K, 3u, 3u);
        __builtin_amdgcn_s_wait_tensorcnt((short)2);  // slab s complete (in-order)
      } else {
        __builtin_amdgcn_s_wait_tensorcnt((short)0);
      }
    }
    __syncthreads();

    const unsigned short* As = &ldsA[buf][0];
    const unsigned short* Bs = &ldsB[buf][0];
    const v16bf a0 = load_a_frag(As, m0l,      LDS_LD, 0, lane);
    const v16bf a1 = load_a_frag(As, m0l + 16, LDS_LD, 0, lane);
#pragma unroll
    for (int j = 0; j < 4; ++j) {
      const v16bf bf = load_b_frag(Bs, n0l + j * 16, LDS_LD, 0, lane);
      acc[0][j] = wmma_bf16(a0, bf, acc[0][j]);
      acc[1][j] = wmma_bf16(a1, bf, acc[1][j]);
    }
    __syncthreads();
  }

  // C/D layout: VGPR r -> row r (lanes 0-15) / r+8 (lanes 16-31), col = lane%16.
  const int half = lane >> 4, nl = lane & 15;
#pragma unroll
  for (int i = 0; i < 2; ++i)
#pragma unroll
    for (int j = 0; j < 4; ++j)
#pragma unroll
      for (int r = 0; r < 8; ++r) {
        const int m = bm * TILE_MN + m0l + i * 16 + r + (half << 3);
        const int n = bn * TILE_MN + n0l + j * 16 + nl;
        float v = acc[i][j][r] + bias[n];
        if (EPI == EPI_BF16) {
          outB[(size_t)m * N + n] = f2bf(v);
        } else if (EPI == EPI_VT) {
          const int bb = m / SEQ_L, l = m - bb * SEQ_L;
          const int hh = n >> 6,   d = n & 63;
          outB[(((size_t)bb * N_HEADS + hh) * D_HEAD + d) * SEQ_L + l] = f2bf(v);
        } else if (EPI == EPI_RES) {
          outF[(size_t)m * N + n] = res[(size_t)m * N + n] + v;
        } else {  // EPI_GELU (exact erf form)
          const float ge = 0.5f * v * (1.0f + erff(v * 0.70710678118654752f));
          outB[(size_t)m * N + n] = f2bf(ge);
        }
      }
}

// ---------------------------------------------------------------------------
// Block-cooperative flash attention. Block = 8 waves = 128 q rows of one
// (b,h). Per 32-key chunk, wave 0 TDM-stages the K tile (32x64, row stride
// D_MODEL) and V^T tile (64x32, row stride SEQ_L) into double-buffered LDS;
// all waves consume via ds_load_b128. Causal: loop to the block's max chunk,
// each wave computes only while c < its own chunk count (wave-uniform branch,
// EXEC stays full around WMMA); barriers are unconditional.
// ---------------------------------------------------------------------------
__global__ void attn_wmma_kernel(const unsigned short* __restrict__ qb,
                                 const unsigned short* __restrict__ kb,
                                 const unsigned short* __restrict__ vt,
                                 unsigned short* __restrict__ ob,
                                 const int* __restrict__ iscp) {
  __shared__ __align__(16) unsigned short ldsK[2][32 * KT_LD];  // 2 x 4608 B
  __shared__ __align__(16) unsigned short ldsV[2][64 * VT_LD];  // 2 x 5120 B
  __shared__ __align__(16) unsigned short ldsP[8][16 * 32];     // 8 KB

  const int lane = threadIdx.x & 31;
  const int wv   = threadIdx.x >> 5;
  const int b = blockIdx.z, h = blockIdx.y;
  const int q0blk = blockIdx.x * 128;
  const int q0 = q0blk + wv * 16;
  const int causal = *iscp;

  const unsigned short* Q  = qb + (size_t)b * SEQ_L * D_MODEL + h * D_HEAD;
  const unsigned short* Kp = kb + (size_t)b * SEQ_L * D_MODEL + h * D_HEAD;
  const unsigned short* Vt = vt + ((size_t)b * N_HEADS + h) * D_HEAD * SEQ_L;

  const v16bf aq0 = load_a_frag(Q, q0, D_MODEL, 0,  lane);
  const v16bf aq1 = load_a_frag(Q, q0, D_MODEL, 32, lane);

  v8f o[4] = {};
  float mrow[8], lrow[8];
#pragma unroll
  for (int r = 0; r < 8; ++r) { mrow[r] = -3.0e38f; lrow[r] = 0.f; }

  const int half = lane >> 4, nl = lane & 15;
  const int nchunk_wave = causal ? ((q0 + 15) / 32 + 1)    : (SEQ_L / 32);
  const int nchunk_blk  = causal ? ((q0blk + 127) / 32 + 1) : (SEQ_L / 32);

  const int wvu = __builtin_amdgcn_readfirstlane((int)(threadIdx.x >> 5));
  if (wvu == 0) {
    tdm_load_2d((unsigned)(uintptr_t)&ldsK[0][0], Kp, D_HEAD, 32,
                (unsigned long long)D_MODEL, 4u, 3u);
    tdm_load_2d((unsigned)(uintptr_t)&ldsV[0][0], Vt, 32, D_HEAD,
                (unsigned long long)SEQ_L, 3u, 3u);
  }

  for (int c = 0; c < nchunk_blk; ++c) {
    const int buf = c & 1;
    if (wvu == 0) {
      if (c + 1 < nchunk_blk) {
        tdm_load_2d((unsigned)(uintptr_t)&ldsK[buf ^ 1][0],
                    Kp + (size_t)(c + 1) * 32 * D_MODEL, D_HEAD, 32,
                    (unsigned long long)D_MODEL, 4u, 3u);
        tdm_load_2d((unsigned)(uintptr_t)&ldsV[buf ^ 1][0],
                    Vt + (size_t)(c + 1) * 32, 32, D_HEAD,
                    (unsigned long long)SEQ_L, 3u, 3u);
        __builtin_amdgcn_s_wait_tensorcnt((short)2);  // chunk c staged
      } else {
        __builtin_amdgcn_s_wait_tensorcnt((short)0);
      }
    }
    __syncthreads();

    if (c < nchunk_wave) {  // wave-uniform: EXEC all-ones inside
      const int k0 = c * 32;
      const unsigned short* Ks = &ldsK[buf][0];
      const unsigned short* Vs = &ldsV[buf][0];
      const v16bf bk00 = load_b_frag(Ks, 0,  KT_LD, 0,  lane);
      const v16bf bk01 = load_b_frag(Ks, 0,  KT_LD, 32, lane);
      const v16bf bk10 = load_b_frag(Ks, 16, KT_LD, 0,  lane);
      const v16bf bk11 = load_b_frag(Ks, 16, KT_LD, 32, lane);
      v8f s0 = {}, s1 = {};
      s0 = wmma_bf16(aq0, bk00, s0); s0 = wmma_bf16(aq1, bk01, s0);
      s1 = wmma_bf16(aq0, bk10, s1); s1 = wmma_bf16(aq1, bk11, s1);

      const bool maskme = causal && (c == nchunk_wave - 1);  // diagonal chunk
      const int key0 = k0 + nl, key1 = k0 + 16 + nl;
#pragma unroll
      for (int r = 0; r < 8; ++r) {
        const int qrow = q0 + r + (half << 3);
        float v0 = s0[r] * 0.125f;   // d_head^-0.5
        float v1 = s1[r] * 0.125f;
        if (maskme) {
          if (key0 > qrow) v0 = -1.0e30f;
          if (key1 > qrow) v1 = -1.0e30f;
        }
        float mx = fmaxf(v0, v1);
        mx = fmaxf(mx, __shfl_xor(mx, 1, 16));
        mx = fmaxf(mx, __shfl_xor(mx, 2, 16));
        mx = fmaxf(mx, __shfl_xor(mx, 4, 16));
        mx = fmaxf(mx, __shfl_xor(mx, 8, 16));
        const float mnew = fmaxf(mrow[r], mx);
        const float corr = __expf(mrow[r] - mnew);
        const float p0 = __expf(v0 - mnew);
        const float p1 = __expf(v1 - mnew);
        float ps = p0 + p1;
        ps += __shfl_xor(ps, 1, 16);
        ps += __shfl_xor(ps, 2, 16);
        ps += __shfl_xor(ps, 4, 16);
        ps += __shfl_xor(ps, 8, 16);
        lrow[r] = lrow[r] * corr + ps;
        mrow[r] = mnew;
#pragma unroll
        for (int t = 0; t < 4; ++t) o[t][r] *= corr;
        const int prow = r + (half << 3);
        ldsP[wv][prow * 32 + nl]      = f2bf(p0);
        ldsP[wv][prow * 32 + 16 + nl] = f2bf(p1);
      }
      // Per-wave LDS round trip: DS ops are in-order within a wave.
      asm volatile("" ::: "memory");
      __builtin_amdgcn_wave_barrier();
      const v16bf ap = load_a_frag(&ldsP[wv][0], 0, 32, 0, lane);
      __builtin_amdgcn_wave_barrier();
      asm volatile("" ::: "memory");

      const v16bf bv0 = load_b_frag(Vs, 0,  VT_LD, 0, lane);
      const v16bf bv1 = load_b_frag(Vs, 16, VT_LD, 0, lane);
      const v16bf bv2 = load_b_frag(Vs, 32, VT_LD, 0, lane);
      const v16bf bv3 = load_b_frag(Vs, 48, VT_LD, 0, lane);
      o[0] = wmma_bf16(ap, bv0, o[0]);
      o[1] = wmma_bf16(ap, bv1, o[1]);
      o[2] = wmma_bf16(ap, bv2, o[2]);
      o[3] = wmma_bf16(ap, bv3, o[3]);
    }
    __syncthreads();
  }

#pragma unroll
  for (int t = 0; t < 4; ++t)
#pragma unroll
    for (int r = 0; r < 8; ++r) {
      const int row = q0 + r + (half << 3);
      const int col = h * D_HEAD + t * 16 + nl;
      ob[((size_t)b * SEQ_L + row) * D_MODEL + col] = f2bf(o[t][r] / lrow[r]);
    }
}

// ---------------------------------------------------------------------------
// Driver
// ---------------------------------------------------------------------------
extern "C" void kernel_launch(void* const* d_in, const int* in_sizes, int n_in,
                              void* d_out, int out_size, void* d_ws, size_t ws_size,
                              hipStream_t stream) {
  (void)in_sizes; (void)n_in; (void)out_size; (void)ws_size;
  const float* x  = (const float*)d_in[0];
  const float* Wq = (const float*)d_in[1];
  const float* bq = (const float*)d_in[2];
  const float* Wk = (const float*)d_in[3];
  const float* bk = (const float*)d_in[4];
  const float* Wv = (const float*)d_in[5];
  const float* bv = (const float*)d_in[6];
  const float* Wo = (const float*)d_in[7];
  const float* bo = (const float*)d_in[8];
  const float* g1 = (const float*)d_in[9];
  const float* b1 = (const float*)d_in[10];
  const float* g2 = (const float*)d_in[11];
  const float* b2 = (const float*)d_in[12];
  const float* W1 = (const float*)d_in[13];
  const float* c1 = (const float*)d_in[14];
  const float* W2 = (const float*)d_in[15];
  const float* c2 = (const float*)d_in[16];
  const int*  isc = (const int*)d_in[17];
  float* out = (float*)d_out;

  // Workspace carve (~90 MB total).
  char* wp = (char*)d_ws;
  auto carve = [&](size_t bytes) -> void* {
    void* p = (void*)wp;
    wp += (bytes + 255) & ~((size_t)255);
    return p;
  };
  unsigned short* xn1 = (unsigned short*)carve((size_t)M_TOT * D_MODEL * 2);
  unsigned short* wqB = (unsigned short*)carve((size_t)D_MODEL * D_MODEL * 2);
  unsigned short* wkB = (unsigned short*)carve((size_t)D_MODEL * D_MODEL * 2);
  unsigned short* wvB = (unsigned short*)carve((size_t)D_MODEL * D_MODEL * 2);
  unsigned short* woB = (unsigned short*)carve((size_t)D_MODEL * D_MODEL * 2);
  unsigned short* w1B = (unsigned short*)carve((size_t)D_FF * D_MODEL * 2);
  unsigned short* w2B = (unsigned short*)carve((size_t)D_FF * D_MODEL * 2);
  unsigned short* qB  = (unsigned short*)carve((size_t)M_TOT * D_MODEL * 2);
  unsigned short* kB  = (unsigned short*)carve((size_t)M_TOT * D_MODEL * 2);
  unsigned short* vtB = (unsigned short*)carve((size_t)M_TOT * D_MODEL * 2);
  unsigned short* atB = (unsigned short*)carve((size_t)M_TOT * D_MODEL * 2);
  float*          x1F = (float*)carve((size_t)M_TOT * D_MODEL * 4);
  unsigned short* xn2 = (unsigned short*)carve((size_t)M_TOT * D_MODEL * 2);
  unsigned short* hB  = (unsigned short*)carve((size_t)M_TOT * D_FF * 2);

  const int nWW = D_MODEL * D_MODEL;   // 589824
  const int nWF = D_FF * D_MODEL;      // 2359296
  cast_bf16_kernel<<<(nWW + 255) / 256, 256, 0, stream>>>(Wq, wqB, nWW);
  cast_bf16_kernel<<<(nWW + 255) / 256, 256, 0, stream>>>(Wk, wkB, nWW);
  cast_bf16_kernel<<<(nWW + 255) / 256, 256, 0, stream>>>(Wv, wvB, nWW);
  cast_bf16_kernel<<<(nWW + 255) / 256, 256, 0, stream>>>(Wo, woB, nWW);
  cast_bf16_kernel<<<(nWF + 255) / 256, 256, 0, stream>>>(W1, w1B, nWF);
  cast_bf16_kernel<<<(nWF + 255) / 256, 256, 0, stream>>>(W2, w2B, nWF);

  // LN1
  ln_bf16_kernel<<<M_TOT, 256, 0, stream>>>(x, g1, b1, xn1);

  // Q, K, V projections (M=4096, N=768, K=768): grid = (N/128, M/128).
  const dim3 gProj(D_MODEL / TILE_MN, M_TOT / TILE_MN);   // (6, 32)
  gemm_wmma_kernel<EPI_BF16><<<gProj, 256, 0, stream>>>(
      xn1, wqB, bq, M_TOT, D_MODEL, D_MODEL, qB, nullptr, nullptr);
  gemm_wmma_kernel<EPI_BF16><<<gProj, 256, 0, stream>>>(
      xn1, wkB, bk, M_TOT, D_MODEL, D_MODEL, kB, nullptr, nullptr);
  gemm_wmma_kernel<EPI_VT><<<gProj, 256, 0, stream>>>(
      xn1, wvB, bv, M_TOT, D_MODEL, D_MODEL, vtB, nullptr, nullptr);

  // Attention
  attn_wmma_kernel<<<dim3(SEQ_L / 128, N_HEADS, BATCH), 256, 0, stream>>>(
      qB, kB, vtB, atB, isc);

  // O projection + residual -> x1 (fp32)
  gemm_wmma_kernel<EPI_RES><<<gProj, 256, 0, stream>>>(
      atB, woB, bo, M_TOT, D_MODEL, D_MODEL, nullptr, x, x1F);

  // LN2
  ln_bf16_kernel<<<M_TOT, 256, 0, stream>>>(x1F, g2, b2, xn2);

  // FFN1 + GELU (M=4096, N=3072, K=768)
  const dim3 gFF1(D_FF / TILE_MN, M_TOT / TILE_MN);       // (24, 32)
  gemm_wmma_kernel<EPI_GELU><<<gFF1, 256, 0, stream>>>(
      xn2, w1B, c1, M_TOT, D_FF, D_MODEL, hB, nullptr, nullptr);

  // FFN2 + residual -> d_out (M=4096, N=768, K=3072)
  gemm_wmma_kernel<EPI_RES><<<gProj, 256, 0, stream>>>(
      hB, w2B, c2, M_TOT, D_MODEL, D_FF, nullptr, x1F, out);
}
